// SpaMambaProcessor_49709951484476
// MI455X (gfx1250) — compile-verified, matrix operand order checked by reading, not run on hardware
//
#include <hip/hip_runtime.h>
#include <hip/hip_bf16.h>
#include <math.h>

// ---------------------------------------------------------------------------
// Bidirectional Mamba (Vision-Mamba) for gfx1250 / MI455X.
//   K0   : f32 -> bf16 (RNE) conversion of GEMM operands
//   K1   : in-proj  XZ = X @ in_w^T      via v_wmma_f32_16x16x32_bf16
//   K2   : depthwise causal conv(4) + SiLU, both scan directions
//   K3   : dbl = xc @ x_w^T (40 outs), dt = softplus(dt8 @ dt_w^T + dt_b)
//   K4a/b/c : CHUNKED selective scan (Blelloch over 196 chunks x 64 steps)
//             -> critical path 64+196+64 instead of 12544 serial steps.
//             BC vectors staged to LDS via global_load_async_to_lds_b128.
//   K5   : sum fwd/bwd -> bf16
//   K6   : out-proj  OUT = Y @ out_w^T   via v_wmma_f32_16x16x32_bf16
// ---------------------------------------------------------------------------

#define D_MODEL 128
#define D_INNER 256
#define D_STATE 16
#define DT_RANK 8
#define BATCH   2
#define SEQLEN  (112 * 112)        // 12544 = 196 * 64
#define MROWS   (BATCH * SEQLEN)   // 25088, multiple of 16
#define CHUNKS  196
#define CSTEPS  64

typedef __attribute__((ext_vector_type(16))) __bf16 v16bf;
typedef __attribute__((ext_vector_type(8)))  float  v8f;

union Frag { uint4 q[2]; v16bf v; };

// ---- gfx1250 async global->LDS path (guarded; falls back to float4 copies) -
#if defined(__has_builtin)
#if __has_builtin(__builtin_amdgcn_global_load_async_to_lds_b128) && \
    __has_builtin(__builtin_amdgcn_s_wait_asynccnt)
#define HAVE_ASYNC_LDS 1
#endif
#endif
#ifndef HAVE_ASYNC_LDS
#define HAVE_ASYNC_LDS 0
#endif

// Builtin signature (from hipcc diagnostic): param0 is a NON-const pointer to
// int __attribute__((vector_size(16))) in the global (__device__/AS1) space.
typedef int async_v4i __attribute__((__vector_size__(16)));
typedef __attribute__((address_space(1))) async_v4i* gbl_v4i_ptr;
typedef __attribute__((address_space(3))) async_v4i* lds_v4i_ptr;

// Stage 8KB (CSTEPS*32 floats = 512 x b128) of BC vectors into LDS, 256 thr.
__device__ __forceinline__ void stage_bc_chunk(const float* g, float* l, int t) {
#if HAVE_ASYNC_LDS
    float* gnc = const_cast<float*>(g);
    gbl_v4i_ptr gp = (gbl_v4i_ptr)(void*)gnc;
    lds_v4i_ptr lp = (lds_v4i_ptr)(void*)l;
    __builtin_amdgcn_global_load_async_to_lds_b128(gp + t,       lp + t,       0, 0);
    __builtin_amdgcn_global_load_async_to_lds_b128(gp + t + 256, lp + t + 256, 0, 0);
    __builtin_amdgcn_s_wait_asynccnt(0);
#else
    float4* dv = (float4*)l;
    const float4* sv = (const float4*)g;
    dv[t]       = sv[t];
    dv[t + 256] = sv[t + 256];
#endif
    __syncthreads();
}

// ---------------------------------------------------------------- K0: cvt
__global__ void f32_to_bf16_rne(const float* __restrict__ s,
                                unsigned short* __restrict__ d, int n) {
    int i = blockIdx.x * blockDim.x + threadIdx.x;
    if (i < n) {
        unsigned u = __float_as_uint(s[i]);
        unsigned r = u + 0x7FFFu + ((u >> 16) & 1u);   // round-to-nearest-even
        d[i] = (unsigned short)(r >> 16);
    }
}

// ----------------------------------------------------- K1/K6: bf16 WMMA GEMM
// C[M,N] = A[M,K] @ B[N,K]^T  (operands row-major over K, bf16; C f32)
// One wave -> 16x64 tile: A-fragment reused across 4 WMMAs per k-step.
__global__ void __launch_bounds__(256)
gemm_bf16_wmma(const unsigned short* __restrict__ A,
               const unsigned short* __restrict__ Bw,
               float* __restrict__ C, int M, int N, int K) {
    const int lane = threadIdx.x & 31;
    const int wave = blockIdx.x * 8 + (threadIdx.x >> 5);
    const int nCols = N >> 6;
    const int tm = wave / nCols;
    const int tn = wave % nCols;
    if (tm * 16 >= M) return;
    const int m0 = tm * 16, n0 = tn * 64;
    // 16-bit A 16x32 layout: lanes 0-15 row=lane, K {0..7,16..23};
    //                        lanes 16-31 row=lane-16, K {8..15,24..31}.
    const int row  = lane & 15;
    const int koff = (lane >> 4) << 3;

    v8f acc[4];
#pragma unroll
    for (int j = 0; j < 4; ++j)
#pragma unroll
        for (int r = 0; r < 8; ++r) acc[j][r] = 0.0f;

    for (int k0 = 0; k0 < K; k0 += 32) {
        Frag fa;
        const unsigned short* ap = A + (size_t)(m0 + row) * K + k0 + koff;
        fa.q[0] = *(const uint4*)(ap);
        fa.q[1] = *(const uint4*)(ap + 16);
#pragma unroll
        for (int j = 0; j < 4; ++j) {
            Frag fb;
            const unsigned short* bp =
                Bw + (size_t)(n0 + j * 16 + row) * K + k0 + koff;
            fb.q[0] = *(const uint4*)(bp);
            fb.q[1] = *(const uint4*)(bp + 16);
            acc[j] = __builtin_amdgcn_wmma_f32_16x16x32_bf16(
                false, fa.v, false, fb.v, (short)0, acc[j], false, false);
        }
    }
    const int rbase = (lane < 16) ? 0 : 8;
    const int col   = lane & 15;
#pragma unroll
    for (int j = 0; j < 4; ++j)
#pragma unroll
        for (int r = 0; r < 8; ++r)
            C[(size_t)(m0 + rbase + r) * N + (n0 + j * 16 + col)] = acc[j][r];
}

// ------------------------------------------------------------ K2: conv+SiLU
__global__ void __launch_bounds__(256)
conv_silu_kernel(const float* __restrict__ xz, const float* __restrict__ conv_w,
                 const float* __restrict__ conv_b, float* __restrict__ xcd,
                 int Bn, int L) {
    const int d   = threadIdx.x;
    const int bl  = blockIdx.x;           // b*L + s
    const int dir = blockIdx.y;
    const int b   = bl / L;
    const int s   = bl - b * L;
    float acc = conv_b[d];
#pragma unroll
    for (int j = 0; j < 4; ++j) {
        int p = s - 3 + j;
        if (p >= 0) {
            int lo = dir ? (L - 1 - p) : p;
            acc += conv_w[d * 4 + j] * xz[((size_t)(b * L + lo)) * 512 + d];
        }
    }
    float sig = 1.0f / (1.0f + __expf(-acc));
    xcd[((size_t)dir * Bn * L + bl) * 256 + d] = acc * sig;
}

// ------------------------------------------------- K3: x-proj + dt softplus
__global__ void __launch_bounds__(128)
dbl_dt_kernel(const float* __restrict__ xcd, const float* __restrict__ x_w,
              const float* __restrict__ dt_w, const float* __restrict__ dt_b,
              float* __restrict__ dtf, float* __restrict__ BC, int Bn, int L) {
    const int dir = blockIdx.y;
    const int bl  = blockIdx.x;
    const int t   = threadIdx.x;
    const float* xrow = xcd + ((size_t)dir * Bn * L + bl) * 256;
    __shared__ float xs[256];
    __shared__ float dbl[DT_RANK + 2 * D_STATE];
    xs[t]       = xrow[t];
    xs[t + 128] = xrow[t + 128];
    __syncthreads();
    if (t < 40) {
        float acc = 0.0f;
        const float* wr = x_w + t * 256;
        for (int k = 0; k < 256; ++k) acc += wr[k] * xs[k];
        dbl[t] = acc;
    }
    __syncthreads();
    for (int d = t; d < 256; d += 128) {
        float acc = dt_b[d];
#pragma unroll
        for (int r = 0; r < DT_RANK; ++r) acc += dt_w[d * DT_RANK + r] * dbl[r];
        float sp = (acc > 20.0f) ? acc : log1pf(__expf(acc));
        dtf[((size_t)dir * Bn * L + bl) * 256 + d] = sp;
    }
    if (t < 32) BC[((size_t)dir * Bn * L + bl) * 32 + t] = dbl[DT_RANK + t];
}

// --------------------------------------------------- K4a: chunk-local scan
// For each (dir,b,chunk,channel): P[n] = prod_a, h_part[n] = local scan(h0=0)
// Layout of P/H buffers: (((dir*Bn+b)*CHUNKS + c)*16 + n)*256 + d  (coalesced)
__global__ void __launch_bounds__(256)
scan_pass1(const float* __restrict__ xcd, const float* __restrict__ dtf,
           const float* __restrict__ BC, const float* __restrict__ A_log,
           float* __restrict__ Pbuf, float* __restrict__ Hbuf, int Bn, int L) {
    const int c = blockIdx.x, b = blockIdx.y, dir = blockIdx.z;
    const int d = threadIdx.x;
    const size_t cbase = (size_t)dir * Bn * L + (size_t)b * L;
    const size_t s0 = (size_t)c * CSTEPS;

    __shared__ float bcs[CSTEPS * 32];
    stage_bc_chunk(BC + (cbase + s0) * 32, bcs, d);

    float Arow[D_STATE], h[D_STATE], P[D_STATE];
#pragma unroll
    for (int n = 0; n < D_STATE; ++n) {
        Arow[n] = -__expf(A_log[d * D_STATE + n]);
        h[n] = 0.0f;
        P[n] = 1.0f;
    }
    const float* dtp = dtf + (cbase + s0) * 256 + d;
    const float* xcp = xcd + (cbase + s0) * 256 + d;
    for (int s = 0; s < CSTEPS; ++s) {
        const float dt = dtp[(size_t)s * 256];
        const float xv = xcp[(size_t)s * 256];
        const float* bm = bcs + s * 32;      // LDS broadcast
#pragma unroll
        for (int n = 0; n < D_STATE; ++n) {
            float a = __expf(dt * Arow[n]);
            h[n] = a * h[n] + (dt * bm[n]) * xv;
            P[n] *= a;
        }
    }
    const size_t obase = ((((size_t)dir * Bn + b) * CHUNKS + c) * 16) * 256 + d;
#pragma unroll
    for (int n = 0; n < D_STATE; ++n) {
        Pbuf[obase + (size_t)n * 256] = P[n];
        Hbuf[obase + (size_t)n * 256] = h[n];
    }
}

// -------------------------------------------- K4b: serial scan over chunks
// Converts Hbuf in-place from h_part(c) into the true initial state of c.
__global__ void __launch_bounds__(256)
scan_pass2(const float* __restrict__ Pbuf, float* __restrict__ Hbuf, int Bn) {
    const int d = threadIdx.x;
    const int b = blockIdx.x, dir = blockIdx.y;
    float h[D_STATE];
#pragma unroll
    for (int n = 0; n < D_STATE; ++n) h[n] = 0.0f;
    for (int c = 0; c < CHUNKS; ++c) {
        const size_t base = ((((size_t)dir * Bn + b) * CHUNKS + c) * 16) * 256 + d;
        __builtin_prefetch(Pbuf + base + 4096, 0, 0);   // global_prefetch_b8
#pragma unroll
        for (int n = 0; n < D_STATE; ++n) {
            float p  = Pbuf[base + (size_t)n * 256];
            float hp = Hbuf[base + (size_t)n * 256];
            Hbuf[base + (size_t)n * 256] = h[n];        // h_init for chunk c
            h[n] = p * h[n] + hp;
        }
    }
}

// ------------------------------------------- K4c: re-scan chunk, emit gated y
__global__ void __launch_bounds__(256)
scan_pass3(const float* __restrict__ xcd, const float* __restrict__ dtf,
           const float* __restrict__ BC, const float* __restrict__ xz,
           const float* __restrict__ A_log, const float* __restrict__ Dv,
           const float* __restrict__ Hbuf, float* __restrict__ Y0,
           float* __restrict__ Y1, int Bn, int L) {
    const int c = blockIdx.x, b = blockIdx.y, dir = blockIdx.z;
    const int d = threadIdx.x;
    float* __restrict__ Y = dir ? Y1 : Y0;
    const size_t cbase = (size_t)dir * Bn * L + (size_t)b * L;
    const size_t s0 = (size_t)c * CSTEPS;

    __shared__ float bcs[CSTEPS * 32];
    stage_bc_chunk(BC + (cbase + s0) * 32, bcs, d);

    float Arow[D_STATE], h[D_STATE];
    const size_t ibase = ((((size_t)dir * Bn + b) * CHUNKS + c) * 16) * 256 + d;
#pragma unroll
    for (int n = 0; n < D_STATE; ++n) {
        Arow[n] = -__expf(A_log[d * D_STATE + n]);
        h[n] = Hbuf[ibase + (size_t)n * 256];          // true initial state
    }
    const float Dd = Dv[d];
    const float* dtp = dtf + (cbase + s0) * 256 + d;
    const float* xcp = xcd + (cbase + s0) * 256 + d;
    for (int s = 0; s < CSTEPS; ++s) {
        const float dt = dtp[(size_t)s * 256];
        const float xv = xcp[(size_t)s * 256];
        const float* bm = bcs + s * 32;
        const float* cm = bcs + s * 32 + 16;
        float y = 0.0f;
#pragma unroll
        for (int n = 0; n < D_STATE; ++n) {
            float a = __expf(dt * Arow[n]);            // v_exp_f32 (TRANS)
            h[n] = a * h[n] + (dt * bm[n]) * xv;
            y += h[n] * cm[n];
        }
        const int gs = (int)s0 + s;
        const int lo = dir ? (L - 1 - gs) : gs;
        const float zv = xz[((size_t)(b * L + lo)) * 512 + 256 + d];
        const float gate = zv / (1.0f + __expf(-zv));  // silu(z)
        Y[((size_t)(b * L + lo)) * 256 + d] = (y + Dd * xv) * gate;
    }
}

// ---------------------------------------------------------- K5: sum -> bf16
__global__ void sum_to_bf16(const float* __restrict__ a,
                            const float* __restrict__ b,
                            unsigned short* __restrict__ o, int n) {
    int i = blockIdx.x * blockDim.x + threadIdx.x;
    if (i < n) {
        unsigned u = __float_as_uint(a[i] + b[i]);
        unsigned r = u + 0x7FFFu + ((u >> 16) & 1u);
        o[i] = (unsigned short)(r >> 16);
    }
}

// ---------------------------------------------------------------------------
extern "C" void kernel_launch(void* const* d_in, const int* in_sizes, int n_in,
                              void* d_out, int out_size, void* d_ws, size_t ws_size,
                              hipStream_t stream) {
    (void)in_sizes; (void)n_in; (void)out_size; (void)ws_size;
    const int Bn = BATCH, L = SEQLEN, M = MROWS;

    const float* x      = (const float*)d_in[0];
    const float* in_w   = (const float*)d_in[1];
    const float* conv_w = (const float*)d_in[2];
    const float* conv_b = (const float*)d_in[3];
    const float* x_w    = (const float*)d_in[4];
    const float* dt_w   = (const float*)d_in[5];
    const float* dt_b   = (const float*)d_in[6];
    const float* A_log  = (const float*)d_in[7];
    const float* Dv     = (const float*)d_in[8];
    const float* out_w  = (const float*)d_in[9];

    char* ws = (char*)d_ws;
    size_t off = 0;
    auto alloc = [&](size_t bytes) -> char* {
        char* p = ws + off;
        off = (off + bytes + 255) & ~(size_t)255;
        return p;
    };
    unsigned short* Xbf    = (unsigned short*)alloc((size_t)M * 128 * 2);
    unsigned short* Winbf  = (unsigned short*)alloc((size_t)512 * 128 * 2);
    unsigned short* Woutbf = (unsigned short*)alloc((size_t)128 * 256 * 2);
    float* XZ  = (float*)alloc((size_t)M * 512 * 4);
    float* XCD = (float*)alloc((size_t)2 * M * 256 * 4);
    float* DT  = (float*)alloc((size_t)2 * M * 256 * 4);
    float* BC  = (float*)alloc((size_t)2 * M * 32 * 4);
    float* Y0f = (float*)alloc((size_t)M * 256 * 4);
    float* Y1f = (float*)alloc((size_t)M * 256 * 4);
    unsigned short* Ybf = (unsigned short*)alloc((size_t)M * 256 * 2);
    float* Pbuf = (float*)alloc((size_t)4 * CHUNKS * 16 * 256 * 4);
    float* Hbuf = (float*)alloc((size_t)4 * CHUNKS * 16 * 256 * 4);

    // K0: operand conversion (RNE) — bf16 WMMA path
    f32_to_bf16_rne<<<(M * 128 + 255) / 256, 256, 0, stream>>>(x, Xbf, M * 128);
    f32_to_bf16_rne<<<(512 * 128 + 255) / 256, 256, 0, stream>>>(in_w, Winbf, 512 * 128);
    f32_to_bf16_rne<<<(128 * 256 + 255) / 256, 256, 0, stream>>>(out_w, Woutbf, 128 * 256);

    // K1: in-projection, M=25088 N=512 K=128
    {
        int waves = (M / 16) * (512 / 64);
        gemm_bf16_wmma<<<(waves + 7) / 8, 256, 0, stream>>>(Xbf, Winbf, XZ, M, 512, 128);
    }
    // K2: depthwise causal conv + SiLU (fwd + flipped)
    conv_silu_kernel<<<dim3(M, 2), 256, 0, stream>>>(XZ, conv_w, conv_b, XCD, Bn, L);
    // K3: per-token x-proj and dt softplus
    dbl_dt_kernel<<<dim3(M, 2), 128, 0, stream>>>(XCD, x_w, dt_w, dt_b, DT, BC, Bn, L);
    // K4: chunked selective scan (depth 64 + 196 + 64 instead of 12544)
    scan_pass1<<<dim3(CHUNKS, Bn, 2), 256, 0, stream>>>(XCD, DT, BC, A_log, Pbuf, Hbuf, Bn, L);
    scan_pass2<<<dim3(Bn, 2), 256, 0, stream>>>(Pbuf, Hbuf, Bn);
    scan_pass3<<<dim3(CHUNKS, Bn, 2), 256, 0, stream>>>(XCD, DT, BC, XZ, A_log, Dv,
                                                        Hbuf, Y0f, Y1f, Bn, L);
    // K5: fwd + bwd -> bf16 for out-proj
    sum_to_bf16<<<(M * 256 + 255) / 256, 256, 0, stream>>>(Y0f, Y1f, Ybf, M * 256);
    // K6: out-projection, M=25088 N=128 K=256 -> d_out
    {
        int waves = (M / 16) * (128 / 64);
        gemm_bf16_wmma<<<(waves + 7) / 8, 256, 0, stream>>>(Ybf, Woutbf, (float*)d_out,
                                                            M, 128, 256);
    }
}
